// HebbianConv2d_30477087932784
// MI455X (gfx1250) — matrix operand order repaired and checked by reading, not compiled, not run on hardware
//
#include <hip/hip_runtime.h>

typedef _Float16 v16h __attribute__((ext_vector_type(16)));
typedef _Float16 v8h  __attribute__((ext_vector_type(8)));
typedef float    v8f  __attribute__((ext_vector_type(8)));
typedef unsigned int v4u __attribute__((ext_vector_type(4)));
typedef int      v8i  __attribute__((ext_vector_type(8)));
typedef int      v4i  __attribute__((ext_vector_type(4)));

#define B_    32
#define CIN   64
#define COUT  128
#define H_    114
#define W_    114
#define K_    3
#define HO    112
#define WO    112
#define EXC   102
#define KDIM  (CIN * K_ * K_)        // 576
#define NKCH  (KDIM / 32)            // 18 wmma chunks
#define XT_W  18                     // x tile width (16 px + K-1)
#define XT_SZ (CIN * K_ * XT_W)      // 3456 floats
#define BP_SZ (KDIM * 16)            // 9216 halves: im2col panel, fragment order
#define NWG   ((B_ * HO * WO) / 16)  // 25088 workgroups

// ---------------------------------------------------------------------------
// Kernel 0: w = weight*sign (f32 + packed f16), zero accumulators
// ---------------------------------------------------------------------------
__global__ void prep_kernel(const float* __restrict__ weight,
                            const float* __restrict__ sign,
                            float* __restrict__ wf, _Float16* __restrict__ wp,
                            float* __restrict__ yx, float* __restrict__ yu,
                            unsigned* __restrict__ maxabs) {
    int i = blockIdx.x * 256 + threadIdx.x;
    if (i < COUT * KDIM) {
        float v = weight[i] * sign[i];
        wf[i] = v;
        wp[i] = (_Float16)v;
        yx[i] = 0.0f;
    }
    if (i < COUT) yu[i] = 0.0f;
    if (i == 0)  *maxabs = 0u;
}

// ---------------------------------------------------------------------------
// Kernel 1: forward conv (WMMA implicit GEMM) + in-block WTA + sparse scatter
//   block = 256 threads = 8 waves; wave s computes channels [16s,16s+16) for
//   the block's 16 contiguous pixels. x tile comes in via the Tensor Data
//   Mover (3D tile DMA -> LDS), B panel built once per block in fragment
//   order so the WMMA loop is pure loads + wmma.
// ---------------------------------------------------------------------------
__global__ __launch_bounds__(256)
void fwd_wta_kernel(const float* __restrict__ x,
                    const _Float16* __restrict__ wp,
                    float* __restrict__ y,
                    float* __restrict__ yx, float* __restrict__ yu) {
    __shared__ float    xs[XT_SZ];     // x patch tile (f32): [ci][kh][18 cols]
    __shared__ _Float16 bp[BP_SZ];     // B panel: [kc][half][lane][8]
    __shared__ int      poff[KDIM];    // k -> patch offset (ci*54 + kh*18 + kw)
    __shared__ float    yt[COUT * 16]; // y tile: 128 ch x 16 px
    __shared__ int      widx[32];
    __shared__ float    wval[32];

    const int n0  = blockIdx.x * 16;
    const int b   = n0 / (HO * WO);
    const int rp  = n0 % (HO * WO);
    const int ho  = rp / WO;
    const int wo0 = rp % WO;           // 112 % 16 == 0 -> tile stays in row

    const int lane = threadIdx.x & 31;
    const int slab = threadIdx.x >> 5; // 0..7 -> wave / channel slab

    // ---- TDM: DMA x[b, 0..63, ho..ho+2, wo0..wo0+17] into xs (wave 0) ----
    if (slab == 0) {
        unsigned long long ga =
            (unsigned long long)(uintptr_t)(x + ((b * CIN) * H_ + ho) * W_ + wo0);
        unsigned int lds_off = (unsigned int)(uintptr_t)(&xs[0]); // low 32 = LDS offset

        v4u g0;
        g0[0] = 1u;                           // count=1, load, no gather
        g0[1] = lds_off;                      // LDS destination
        g0[2] = (unsigned int)ga;             // global_addr[31:0]
        g0[3] = (unsigned int)((ga >> 32) & 0x1FFFFFFu) | (2u << 30); // [56:32]|type=2

        const unsigned long long s0 = W_;            // dim0 stride (elements)
        const unsigned long long s1 = H_ * W_;       // dim1 stride (elements)
        v8i g1;
        g1[0] = (int)(2u << 16);                     // wg_mask=0, data_size=4B
        g1[1] = (int)((W_ & 0xFFFFu) << 16);         // tensor_dim0[15:0]
        g1[2] = (int)((W_ >> 16) | ((H_ & 0xFFFFu) << 16));  // dim0 hi | dim1 lo
        g1[3] = (int)((H_ >> 16) | (XT_W << 16));    // dim1 hi | tile_dim0=18
        g1[4] = (int)(K_ | (CIN << 16));             // tile_dim1=3 | tile_dim2=64
        g1[5] = (int)(s0 & 0xFFFFFFFFu);             // dim0_stride lo
        g1[6] = (int)(((s0 >> 32) & 0xFFFFu) | ((s1 & 0xFFFFu) << 16));
        g1[7] = (int)(s1 >> 16);                     // dim1_stride[47:16]
        v4i g2;
        g2[0] = CIN;                                 // tensor_dim2
        g2[1] = 0; g2[2] = 0; g2[3] = 0;             // 3D tile: rest unused
        v4i g3 = {0, 0, 0, 0};
        v8i g4 = {0, 0, 0, 0, 0, 0, 0, 0};           // unused group (6-arg form)

        __builtin_amdgcn_tensor_load_to_lds(g0, g1, g2, g3, g4, 0);
    }

    // ---- k -> patch-offset table (overlaps with TDM transfer) ----
    for (int k = threadIdx.x; k < KDIM; k += 256) {
        int ci = k / 9;
        int r  = k - ci * 9;
        int kh = r / 3;
        int kw = r - kh * 3;
        poff[k] = ci * (K_ * XT_W) + kh * XT_W + kw;
    }
    if (slab == 0) __builtin_amdgcn_s_wait_tensorcnt(0);
    __syncthreads();

    // ---- build im2col B panel in WMMA fragment order (once per block) ----
    for (int e = threadIdx.x; e < BP_SZ; e += 256) {
        int k   = e >> 4;
        int px  = e & 15;
        int kc  = k >> 5;
        int rem = k & 31;
        int hi  = rem >> 4;
        int j   = rem & 15;
        int blk = (kc * 2 + (j >> 3)) * 32 + hi * 16 + px;
        bp[blk * 8 + (j & 7)] = (_Float16)xs[poff[k] + px];
    }
    __syncthreads();

    const int px = lane & 15;        // pixel within tile / A-row
    const int hi = lane >> 4;        // lane half (contiguous K split: k = 16*hi + s)

    const _Float16* wrow = wp + (slab * 16 + px) * KDIM;
    const v8h* bpv = (const v8h*)bp;

    v8f acc = {};
#pragma unroll
    for (int kc = 0; kc < NKCH; ++kc) {
        // A: weights 16x32 f16, one 32B clause of global loads (L2-resident)
        int abase = kc * 32 + hi * 16;
        v8h a_lo = *(const v8h*)(wrow + abase);
        v8h a_hi = *(const v8h*)(wrow + abase + 8);
        // B: prebuilt fragment, two 16B LDS loads, zero address math
        v8h b_lo = bpv[(kc * 2 + 0) * 32 + lane];
        v8h b_hi = bpv[(kc * 2 + 1) * 32 + lane];
        v16h a, bm;
#pragma unroll
        for (int j = 0; j < 8; ++j) {
            a[j]  = a_lo[j];  a[8 + j]  = a_hi[j];
            bm[j] = b_lo[j];  bm[8 + j] = b_hi[j];
        }
        acc = __builtin_amdgcn_wmma_f32_16x16x32_f16(
                  false, a, false, bm, (short)0, acc, false, false);
    }

    // ---- write y (global) + y tile (LDS), per ISA C-layout ----
#pragma unroll
    for (int i = 0; i < 8; ++i) {
        int   co = slab * 16 + i + (hi ? 8 : 0);
        float v  = acc[i];
        yt[co * 16 + px] = v;
        y[((b * COUT + co) * HO + ho) * WO + wo0 + px] = v;
    }
    __syncthreads();

    // ---- WTA per pixel: threads 0-15 exc argmax, threads 16-31 inh abs-argmax
    if (threadIdx.x < 32) {
        int p = threadIdx.x & 15;
        if (threadIdx.x < 16) {
            float bestv = yt[p];
            int   besti = 0;
            for (int co = 1; co < EXC; ++co) {
                float v = yt[co * 16 + p];
                if (v > bestv) { bestv = v; besti = co; }
            }
            widx[2 * p] = besti;  wval[2 * p] = bestv;
        } else {
            float v0    = yt[EXC * 16 + p];
            float besta = fabsf(v0);
            float bv    = v0;
            int   bi    = EXC;
            for (int co = EXC + 1; co < COUT; ++co) {
                float v = yt[co * 16 + p];
                float a = fabsf(v);
                if (a > besta) { besta = a; bv = v; bi = co; }
            }
            widx[2 * p + 1] = bi;  wval[2 * p + 1] = bv;
        }
    }
    __syncthreads();

    // ---- sparse scatter: yx[co, :] += v * x_patch ; yu[co] += v ----
    int p   = threadIdx.x >> 3;   // 0..31 -> (pixel, winner) pair
    int sub = threadIdx.x & 7;
    int co  = widx[p];
    float v = wval[p];
    int ppx = p >> 1;
    for (int k = sub; k < KDIM; k += 8) {
        atomicAdd(&yx[co * KDIM + k], v * xs[poff[k] + ppx]);
    }
    if (sub == 0) atomicAdd(&yu[co], v);
}

// ---------------------------------------------------------------------------
// Kernel 2: u = yx - yu*w (in place), global max|u| via uint atomicMax
// ---------------------------------------------------------------------------
__global__ void finalize1_kernel(float* __restrict__ yx,
                                 const float* __restrict__ yu,
                                 const float* __restrict__ wf,
                                 unsigned* __restrict__ maxabs) {
    __shared__ float red[256];
    int i = blockIdx.x * 256 + threadIdx.x;
    float u = 0.0f;
    if (i < COUT * KDIM) {
        u = yx[i] - yu[i / KDIM] * wf[i];
        yx[i] = u;
    }
    red[threadIdx.x] = fabsf(u);
    __syncthreads();
    for (int s = 128; s > 0; s >>= 1) {
        if (threadIdx.x < s)
            red[threadIdx.x] = fmaxf(red[threadIdx.x], red[threadIdx.x + s]);
        __syncthreads();
    }
    if (threadIdx.x == 0)
        atomicMax(maxabs, __float_as_uint(red[0]));  // nonneg f32: uint-monotone
}

// ---------------------------------------------------------------------------
// Kernel 3: normalize update into output tail
// ---------------------------------------------------------------------------
__global__ void finalize2_kernel(const float* __restrict__ yx,
                                 const unsigned* __restrict__ maxabs,
                                 float* __restrict__ out) {
    int i = blockIdx.x * 256 + threadIdx.x;
    if (i >= COUT * KDIM) return;
    float m = __uint_as_float(*maxabs);
    out[i] = yx[i] / (m + 1e-8f);
}

// ---------------------------------------------------------------------------
extern "C" void kernel_launch(void* const* d_in, const int* in_sizes, int n_in,
                              void* d_out, int out_size, void* d_ws, size_t ws_size,
                              hipStream_t stream) {
    const float* x      = (const float*)d_in[0];
    const float* weight = (const float*)d_in[1];
    const float* sign   = (const float*)d_in[2];

    float* y       = (float*)d_out;
    float* upd_out = y + (size_t)B_ * COUT * HO * WO;

    char* ws = (char*)d_ws;
    float*    wf     = (float*)ws;                                   // 73728 f32
    _Float16* wp     = (_Float16*)(ws + (size_t)COUT * KDIM * 4);    // 73728 f16
    float*    yx     = (float*)(ws + (size_t)COUT * KDIM * 6);       // 73728 f32
    float*    yu     = yx + COUT * KDIM;                             // 128 f32
    unsigned* maxabs = (unsigned*)(yu + COUT);

    prep_kernel<<<(COUT * KDIM + 255) / 256, 256, 0, stream>>>(
        weight, sign, wf, wp, yx, yu, maxabs);
    fwd_wta_kernel<<<NWG, 256, 0, stream>>>(x, wp, y, yx, yu);
    finalize1_kernel<<<(COUT * KDIM + 255) / 256, 256, 0, stream>>>(
        yx, yu, wf, maxabs);
    finalize2_kernel<<<(COUT * KDIM + 255) / 256, 256, 0, stream>>>(
        yx, maxabs, upd_out);
}
